// MoE_83846351553324
// MI455X (gfx1250) — matrix operand order repaired and checked by reading, not compiled, not run on hardware
//
#include <hip/hip_runtime.h>

#define D_MODEL 1024
#define D_FF    4096
#define NE      8
#define T_TOK   8192
#define MT      64      // tokens per tile
#define FFC     512     // d_ff chunk held in LDS
#define THREADS 512     // 16 waves

typedef unsigned int   u32;
typedef unsigned short u16;

typedef __attribute__((ext_vector_type(16))) __bf16 bf16x16;
typedef __attribute__((ext_vector_type(8)))  float  f32x8;

union FragU {
    u32     u[8];
    uint4   q[2];
    bf16x16 v;
};

__device__ __forceinline__ u16 f2bf(float f) {
    u32 u = __float_as_uint(f);
    u32 r = (u + 0x7FFFu + ((u >> 16) & 1u)) >> 16;   // round-to-nearest-even
    return (u16)r;
}

// ---------------------------------------------------------------------------
// Gating: one wave32 per token. Computes gate logits, adds gumbel, top-2 +
// softmax, appends (token, weight) to per-expert bucket lists.
// ---------------------------------------------------------------------------
__global__ void gate_kernel(const float* __restrict__ x,
                            const float* __restrict__ gumbel,
                            const float* __restrict__ gw,
                            const float* __restrict__ gb,
                            int* __restrict__ cnt,
                            int* __restrict__ tok,
                            float* __restrict__ wt)
{
    int t    = (blockIdx.x * blockDim.x + threadIdx.x) >> 5;   // token id
    int lane = threadIdx.x & 31;
    if (t >= T_TOK) return;

    const float* xr = x + (size_t)t * D_MODEL;
    float acc[NE];
#pragma unroll
    for (int e = 0; e < NE; ++e) acc[e] = 0.f;

    for (int d = lane; d < D_MODEL; d += 32) {
        float xv = xr[d];
        const float* g = gw + d * NE;
#pragma unroll
        for (int e = 0; e < NE; ++e) acc[e] += xv * g[e];
    }
#pragma unroll
    for (int off = 16; off >= 1; off >>= 1) {
#pragma unroll
        for (int e = 0; e < NE; ++e) acc[e] += __shfl_xor(acc[e], off, 32);
    }

    if (lane == 0) {
        float v[NE];
#pragma unroll
        for (int e = 0; e < NE; ++e) v[e] = acc[e] + gb[e] + gumbel[t * NE + e];
        int   i1 = 0, i2 = -1;
        float m1 = -INFINITY, m2 = -INFINITY;
#pragma unroll
        for (int e = 0; e < NE; ++e) {
            float val = v[e];
            if (val > m1)       { m2 = m1; i2 = i1; m1 = val; i1 = e; }
            else if (val > m2)  { m2 = val; i2 = e; }
        }
        float e2  = __expf(m2 - m1);
        float wa  = 1.f / (1.f + e2);
        float wb  = e2  / (1.f + e2);
        int p1 = atomicAdd(&cnt[i1], 1);
        tok[i1 * T_TOK + p1] = t;  wt[i1 * T_TOK + p1] = wa;
        int p2 = atomicAdd(&cnt[i2], 1);
        tok[i2 * T_TOK + p2] = t;  wt[i2 * T_TOK + p2] = wb;
    }
}

// ---------------------------------------------------------------------------
// Transpose+convert: src f32 [E][K][N] -> dst bf16 [E][N][K]
// grid (N/32, K/32, E), block (32, 8)
// ---------------------------------------------------------------------------
__global__ void transpose_bf16_kernel(const float* __restrict__ src,
                                      u16* __restrict__ dst, int K, int N)
{
    __shared__ float tile[32][33];
    int e  = blockIdx.z;
    const float* s = src + (size_t)e * K * N;
    u16*         d = dst + (size_t)e * K * N;
    int n0 = blockIdx.x * 32, k0 = blockIdx.y * 32;
    int tx = threadIdx.x, ty = threadIdx.y;
#pragma unroll
    for (int j = 0; j < 32; j += 8)
        tile[ty + j][tx] = s[(size_t)(k0 + ty + j) * N + n0 + tx];
    __syncthreads();
#pragma unroll
    for (int j = 0; j < 32; j += 8)
        d[(size_t)(n0 + ty + j) * K + k0 + tx] = f2bf(tile[tx][ty + j]);
}

// ---------------------------------------------------------------------------
// Fragment loaders (contiguous k order, same for A and B, wave32)
// ---------------------------------------------------------------------------
// A 16x32 bf16: lane l -> row m=l&15; k-base = (l>>4)*16; VGPR0-7 pack k+0..15
// -> two contiguous 16B loads at +0 and +8 elements.
__device__ __forceinline__ bf16x16 load_frag_a(const u16* p) {
    FragU f;
    f.q[0] = *(const uint4*)(p);
    f.q[1] = *(const uint4*)(p + 8);
    return f.v;
}
// B 32x16 bf16 from [N][K] layout: lane l -> col n=l&15, k-base=(l>>4)*16;
// 16 contiguous k values -> two contiguous 16B loads at +0 and +8 elements.
__device__ __forceinline__ bf16x16 load_frag_b_bf16(const u16* p) {
    FragU f;
    f.q[0] = *(const uint4*)(p);
    f.q[1] = *(const uint4*)(p + 8);
    return f.v;
}
// Fallback: B from original f32 [K][N] layout (no workspace for transposed wts)
__device__ __forceinline__ bf16x16 load_frag_b_f32(const float* w, int n,
                                                   int koff, int ldN) {
    FragU f;
#pragma unroll
    for (int j = 0; j < 8; ++j) {
        float lo = w[(size_t)(koff + 2 * j)     * ldN + n];
        float hi = w[(size_t)(koff + 2 * j + 1) * ldN + n];
        f.u[j] = ((u32)f2bf(hi) << 16) | (u32)f2bf(lo);
    }
    return f.v;
}

// ---------------------------------------------------------------------------
// Fused expert FFN: one workgroup = (expert, 64-token tile), 16 waves.
// Wave w: mg = w>>3 owns 32 rows (2 WMMA m-tiles), ng = w&7 owns a col strip:
//   GEMM1: 64 of 512 d_ff cols (4 n-tiles); GEMM2: 128 of 1024 out cols (8).
// Dynamic LDS: xs 128KB + hs 64KB + token metadata (~192.5 KB of 320 KB/WGP).
// All fragment addresses are (hoisted per-lane base) + compile-time-constant
// offsets so the backend folds them into ds/global instruction offsets.
// ---------------------------------------------------------------------------
template <bool TW>
__global__ __launch_bounds__(THREADS)
void moe_ffn_kernel(const float* __restrict__ x,
                    const float* __restrict__ w1f, const float* __restrict__ b1,
                    const float* __restrict__ w2f, const float* __restrict__ b2,
                    const u16* __restrict__ w1t, const u16* __restrict__ w2t,
                    const int* __restrict__ cnt, const int* __restrict__ tok,
                    const float* __restrict__ wt, float* __restrict__ out)
{
    extern __shared__ char smem[];
    u16*   xs    = (u16*)smem;                                   // [MT][D_MODEL]
    u16*   hs    = (u16*)(smem + (size_t)MT * D_MODEL * 2);      // [MT][FFC]
    int*   s_tok = (int*)(smem + (size_t)MT * D_MODEL * 2 + (size_t)MT * FFC * 2);
    float* s_wt  = (float*)(s_tok + MT);

    const int e  = blockIdx.y;
    const int m0 = blockIdx.x * MT;
    const int ce = cnt[e];
    if (m0 >= ce) return;                    // beyond this expert's row count

    const int tid = threadIdx.x;
    if (tid < MT) {
        int slot = m0 + tid;
        if (slot < ce) { s_tok[tid] = tok[e * T_TOK + slot];
                         s_wt[tid]  = wt [e * T_TOK + slot]; }
        else           { s_tok[tid] = 0; s_wt[tid] = 0.f; }
    }
    __syncthreads();

    // gather + convert x rows into LDS (bf16)
    for (int i = tid; i < MT * D_MODEL / 4; i += THREADS) {
        int m = (i * 4) >> 10;
        int d = (i * 4) & (D_MODEL - 1);
        float4 v = *(const float4*)(x + (size_t)s_tok[m] * D_MODEL + d);
        u32 lo = ((u32)f2bf(v.y) << 16) | (u32)f2bf(v.x);
        u32 hi = ((u32)f2bf(v.w) << 16) | (u32)f2bf(v.z);
        *(uint2*)&xs[m * D_MODEL + d] = make_uint2(lo, hi);
    }
    __syncthreads();

    const int w  = tid >> 5;
    const int l  = tid & 31;
    const int lr = l & 15;      // row (A) / col (B,C)
    const int lh = l >> 4;      // k-half / m-half
    const int mg = w >> 3;      // row-group: rows [mg*32, mg*32+32)
    const int ng = w & 7;       // col-group

    // hoisted per-lane fragment base pointers
    const u16* xsp = xs + (mg * 32 + lr) * D_MODEL + lh * 16;
    const u16* hsp = hs + (mg * 32 + lr) * FFC     + lh * 16;
    const u16* w1p = TW ? w1t + ((size_t)e * D_FF    + ng *  64 + lr) * D_MODEL + lh * 16 : nullptr;
    const u16* w2p = TW ? w2t + ((size_t)e * D_MODEL + ng * 128 + lr) * D_FF    + lh * 16 : nullptr;

    f32x8 zero = {};
    f32x8 acc2[16];             // [nt][mt] -> acc2[nt*2+mt]
#pragma unroll
    for (int i = 0; i < 16; ++i) acc2[i] = zero;

    const float* b1e = b1 + e * D_FF;

    for (int c = 0; c < D_FF / FFC; ++c) {
        // ---- GEMM1: h_chunk = relu(x @ w1[:, chunk] + b1) ----
        f32x8 acc1[8];          // [nt][mt] -> acc1[nt*2+mt]
#pragma unroll
        for (int i = 0; i < 8; ++i) acc1[i] = zero;

        const u16* wp = TW ? w1p + (size_t)c * FFC * D_MODEL : nullptr;
        for (int kb = 0; kb < D_MODEL / 32; ++kb) {
            bf16x16 a0 = load_frag_a(xsp + kb * 32);
            bf16x16 a1 = load_frag_a(xsp + 16 * D_MODEL + kb * 32);
#pragma unroll
            for (int nt = 0; nt < 4; ++nt) {
                bf16x16 b;
                if constexpr (TW)
                    b = load_frag_b_bf16(wp + nt * (16 * D_MODEL) + kb * 32);
                else
                    b = load_frag_b_f32(w1f + (size_t)e * D_MODEL * D_FF,
                                        c * FFC + ng * 64 + nt * 16 + lr,
                                        kb * 32 + lh * 16, D_FF);
                acc1[nt * 2 + 0] = __builtin_amdgcn_wmma_f32_16x16x32_bf16(
                    false, a0, false, b, (short)0, acc1[nt * 2 + 0], false, false);
                acc1[nt * 2 + 1] = __builtin_amdgcn_wmma_f32_16x16x32_bf16(
                    false, a1, false, b, (short)0, acc1[nt * 2 + 1], false, false);
            }
        }
        // bias + relu -> bf16 LDS
#pragma unroll
        for (int nt = 0; nt < 4; ++nt) {
            int ncl  = ng * 64 + nt * 16 + lr;
            float bs = b1e[c * FFC + ncl];
#pragma unroll
            for (int mt = 0; mt < 2; ++mt) {
#pragma unroll
                for (int r = 0; r < 8; ++r) {
                    int m   = mg * 32 + mt * 16 + r + 8 * lh;
                    float h = acc1[nt * 2 + mt][r] + bs;
                    hs[m * FFC + ncl] = f2bf(h > 0.f ? h : 0.f);
                }
            }
        }
        __syncthreads();

        // ---- GEMM2 partial: out += h_chunk @ w2[chunk, :] ----
        const u16* wq = TW ? w2p + c * FFC : nullptr;
        for (int kb = 0; kb < FFC / 32; ++kb) {
            bf16x16 a0 = load_frag_a(hsp + kb * 32);
            bf16x16 a1 = load_frag_a(hsp + 16 * FFC + kb * 32);
#pragma unroll
            for (int nt = 0; nt < 8; ++nt) {
                bf16x16 b;
                if constexpr (TW)
                    b = load_frag_b_bf16(wq + nt * (16 * D_FF) + kb * 32);
                else
                    b = load_frag_b_f32(w2f + (size_t)e * D_FF * D_MODEL,
                                        ng * 128 + nt * 16 + lr,
                                        c * FFC + kb * 32 + lh * 16, D_MODEL);
                acc2[nt * 2 + 0] = __builtin_amdgcn_wmma_f32_16x16x32_bf16(
                    false, a0, false, b, (short)0, acc2[nt * 2 + 0], false, false);
                acc2[nt * 2 + 1] = __builtin_amdgcn_wmma_f32_16x16x32_bf16(
                    false, a1, false, b, (short)0, acc2[nt * 2 + 1], false, false);
            }
        }
        __syncthreads();
    }

    // ---- epilogue: scale by routing weight, scatter-add to output ----
#pragma unroll
    for (int nt = 0; nt < 8; ++nt) {
        int n    = ng * 128 + nt * 16 + lr;
        float bs = b2[e * D_MODEL + n];
#pragma unroll
        for (int mt = 0; mt < 2; ++mt) {
#pragma unroll
            for (int r = 0; r < 8; ++r) {
                int   m  = mg * 32 + mt * 16 + r + 8 * lh;
                float wv = s_wt[m];
                if (wv != 0.f)
                    atomicAdd(out + (size_t)s_tok[m] * D_MODEL + n,
                              (acc2[nt * 2 + mt][r] + bs) * wv);
            }
        }
    }
}

// ---------------------------------------------------------------------------
extern "C" void kernel_launch(void* const* d_in, const int* in_sizes, int n_in,
                              void* d_out, int out_size, void* d_ws, size_t ws_size,
                              hipStream_t stream)
{
    const float* x      = (const float*)d_in[0];
    const float* gumbel = (const float*)d_in[1];
    const float* gate_w = (const float*)d_in[2];
    const float* gate_b = (const float*)d_in[3];
    const float* w1     = (const float*)d_in[4];
    const float* b1     = (const float*)d_in[5];
    const float* w2     = (const float*)d_in[6];
    const float* b2     = (const float*)d_in[7];
    float* out = (float*)d_out;

    char* ws  = (char*)d_ws;
    int*   cnt = (int*)ws;                                   // 32 B (pad 4 KB)
    int*   tok = (int*)(ws + 4096);                          // 256 KB
    float* wtb = (float*)(ws + 4096 + NE * T_TOK * 4);       // 256 KB
    size_t wOff = 1u << 20;
    u16* w1t = (u16*)(ws + wOff);
    u16* w2t = (u16*)(ws + wOff + (size_t)NE * D_MODEL * D_FF * 2);
    size_t need = wOff + (size_t)NE * D_MODEL * D_FF * 4;    // ~129 MB
    bool tw = (ws_size >= need);

    hipMemsetAsync(cnt, 0, 64, stream);
    hipMemsetAsync(out, 0, (size_t)T_TOK * D_MODEL * sizeof(float), stream);

    gate_kernel<<<T_TOK / 8, 256, 0, stream>>>(x, gumbel, gate_w, gate_b,
                                               cnt, tok, wtb);

    const size_t smem_bytes = (size_t)MT * D_MODEL * 2   // xs
                            + (size_t)MT * FFC * 2       // hs
                            + (size_t)MT * 8;            // s_tok + s_wt

    if (tw) {
        transpose_bf16_kernel<<<dim3(D_FF / 32, D_MODEL / 32, NE), dim3(32, 8),
                                0, stream>>>(w1, w1t, D_MODEL, D_FF);
        transpose_bf16_kernel<<<dim3(D_MODEL / 32, D_FF / 32, NE), dim3(32, 8),
                                0, stream>>>(w2, w2t, D_FF, D_MODEL);
        moe_ffn_kernel<true><<<dim3(T_TOK / MT, NE), THREADS, smem_bytes, stream>>>(
            x, w1, b1, w2, b2, w1t, w2t, cnt, tok, wtb, out);
    } else {
        moe_ffn_kernel<false><<<dim3(T_TOK / MT, NE), THREADS, smem_bytes, stream>>>(
            x, w1, b1, w2, b2, nullptr, nullptr, cnt, tok, wtb, out);
    }
}
